// Dependency_GCN_9603546874482
// MI455X (gfx1250) — matrix operand order repaired and checked by reading, not compiled, not run on hardware
//
#include <hip/hip_runtime.h>

#define NN 30000
#define NE 150000
#define NR 10
#define DIM 256
#define KSTEPS 8          // 256 / 32
#define NTILES 16         // 256 / 16
#define LDSPAD 264        // 256 + 8 bf16 pad -> 528B row stride (132 dwords, %64 != 0)

typedef __attribute__((ext_vector_type(16))) __bf16 v16bf;
typedef __attribute__((ext_vector_type(8)))  float  v8f;

union BF16x16 { v16bf v; uint4 q[2]; };

// ---------------- conversion kernels ----------------
__global__ __launch_bounds__(256) void cvt_x_bf16(const float* __restrict__ src,
                                                  __bf16* __restrict__ dst, int n) {
  int i = blockIdx.x * blockDim.x + threadIdx.x;
  if (i < n) dst[i] = (__bf16)src[i];
}

// W[mat][k][n] (row-major) -> Wt[mat][n][k] bf16 (transposed)
__global__ __launch_bounds__(256) void cvt_w_transpose(const float* __restrict__ src,
                                                       __bf16* __restrict__ dst, int nmat) {
  int i = blockIdx.x * blockDim.x + threadIdx.x;
  if (i < nmat * DIM * DIM) {
    int mat = i >> 16;
    int rem = i & 65535;
    int k = rem >> 8;
    int n = rem & 255;
    dst[(size_t)mat * DIM * DIM + (size_t)n * DIM + k] = (__bf16)src[i];
  }
}

// ---------------- edge bucketing by relation ----------------
__global__ __launch_bounds__(256) void rel_hist(const int* __restrict__ rel,
                                                int* __restrict__ cnt, int E) {
  int i = blockIdx.x * blockDim.x + threadIdx.x;
  if (i < E) atomicAdd(&cnt[rel[i]], 1);
}

__global__ void rel_scan(const int* __restrict__ cnt, int* __restrict__ off,
                         int* __restrict__ toff) {
  int o = 0, t = 0;
  for (int r = 0; r < NR; ++r) {
    off[r] = o; toff[r] = t;
    o += cnt[r];
    t += (cnt[r] + 15) >> 4;
  }
  off[NR] = o; toff[NR] = t;
}

__global__ __launch_bounds__(256) void rel_fill(const int* __restrict__ rel,
                                                const int* __restrict__ off,
                                                int* __restrict__ cur,
                                                int* __restrict__ perm, int E) {
  int i = blockIdx.x * blockDim.x + threadIdx.x;
  if (i < E) {
    int r = rel[i];
    int p = off[r] + atomicAdd(&cur[r], 1);
    perm[p] = i;
  }
}

// ---------------- self transform: out = x @ W_self + b_self ----------------
// grid.x = NN/16 workgroups of 256 threads (8 waves); wave w does n-tiles {w, w+8}
__global__ __launch_bounds__(256) void gemm_self(const __bf16* __restrict__ xb,
                                                 const __bf16* __restrict__ Wt,
                                                 const float* __restrict__ bias,
                                                 float* __restrict__ out) {
  __shared__ __bf16 As[16][LDSPAD];
  int tid = threadIdx.x;
  int m0 = blockIdx.x * 16;

  // cooperative stage: 16 rows x 32 uint4 (512B/row)
  for (int c = tid; c < 512; c += 256) {
    int row = c >> 5, q = c & 31;
    *(uint4*)&As[row][q * 8] = *(const uint4*)(xb + (size_t)(m0 + row) * DIM + q * 8);
  }
  __syncthreads();

  int wave = tid >> 5, lane = tid & 31;
  int h = lane >> 4;        // lane half
  int lr = lane & 15;       // row (A) / column (B,C)

  #pragma unroll
  for (int it = 0; it < 2; ++it) {
    int nt = wave + it * 8;
    int col = nt * 16 + lr;
    v8f acc = {};
    #pragma unroll
    for (int ks = 0; ks < KSTEPS; ++ks) {
      BF16x16 a, b;
      int ka = ks * 32 + h * 8;
      a.q[0] = *(const uint4*)&As[lr][ka];
      a.q[1] = *(const uint4*)&As[lr][ka + 16];
      const __bf16* bp = Wt + (size_t)col * DIM + ks * 32 + h * 8;
      b.q[0] = *(const uint4*)(bp);
      b.q[1] = *(const uint4*)(bp + 16);
      acc = __builtin_amdgcn_wmma_f32_16x16x32_bf16(false, a.v, false, b.v,
                                                    (short)0, acc, false, false);
    }
    float bv = bias[col];
    #pragma unroll
    for (int v = 0; v < 8; ++v) {
      int m = m0 + v + h * 8;
      out[(size_t)m * DIM + col] = acc[v] + bv;
    }
  }
}

// ---------------- edge messages: out[sidx[e]] += x[gidx[e]] @ W_r + b_r ----------------
// one workgroup = one 16-edge tile of one relation
__global__ __launch_bounds__(256) void gemm_edge(const __bf16* __restrict__ xb,
                                                 const __bf16* __restrict__ Wt_all,
                                                 const float* __restrict__ bias_all,
                                                 const int* __restrict__ perm,
                                                 const int* __restrict__ gidx,
                                                 const int* __restrict__ sidx,
                                                 const int* __restrict__ off,
                                                 const int* __restrict__ toff,
                                                 float* __restrict__ out) {
  __shared__ __bf16 As[16][LDSPAD];
  __shared__ int s_gather[16];
  __shared__ int s_scatter[16];

  int tile = blockIdx.x;
  if (tile >= toff[NR]) return;   // uniform across block

  // locate relation (10 buckets -> trivial scan)
  int r = 0;
  while (tile >= toff[r + 1]) ++r;
  int base = off[r] + (tile - toff[r]) * 16;
  int end  = off[r + 1];

  int tid = threadIdx.x;
  if (tid < 16) {
    int p = base + tid;
    if (p < end) {
      int e = perm[p];
      s_gather[tid]  = gidx[e];
      s_scatter[tid] = sidx[e];
    } else {
      s_gather[tid]  = -1;
      s_scatter[tid] = -1;
    }
  }
  __syncthreads();

  // gather 16 x-rows into LDS (zero-fill invalid tail slots)
  for (int c = tid; c < 512; c += 256) {
    int row = c >> 5, q = c & 31;
    int g = s_gather[row];
    uint4 val = make_uint4(0u, 0u, 0u, 0u);
    if (g >= 0) val = *(const uint4*)(xb + (size_t)g * DIM + q * 8);
    *(uint4*)&As[row][q * 8] = val;
  }
  __syncthreads();

  const __bf16* Wt  = Wt_all  + (size_t)r * DIM * DIM;
  const float* bias = bias_all + (size_t)r * DIM;

  int wave = tid >> 5, lane = tid & 31;
  int h = lane >> 4;
  int lr = lane & 15;

  #pragma unroll
  for (int it = 0; it < 2; ++it) {
    int nt = wave + it * 8;
    int col = nt * 16 + lr;
    v8f acc = {};
    #pragma unroll
    for (int ks = 0; ks < KSTEPS; ++ks) {
      BF16x16 a, b;
      int ka = ks * 32 + h * 8;
      a.q[0] = *(const uint4*)&As[lr][ka];
      a.q[1] = *(const uint4*)&As[lr][ka + 16];
      const __bf16* bp = Wt + (size_t)col * DIM + ks * 32 + h * 8;
      b.q[0] = *(const uint4*)(bp);
      b.q[1] = *(const uint4*)(bp + 16);
      acc = __builtin_amdgcn_wmma_f32_16x16x32_bf16(false, a.v, false, b.v,
                                                    (short)0, acc, false, false);
    }
    float bv = bias[col];
    #pragma unroll
    for (int v = 0; v < 8; ++v) {
      int dst = s_scatter[v + h * 8];
      if (dst >= 0)
        atomicAdd(&out[(size_t)dst * DIM + col], acc[v] + bv);
    }
  }
}

// ---------------- launch ----------------
extern "C" void kernel_launch(void* const* d_in, const int* in_sizes, int n_in,
                              void* d_out, int out_size, void* d_ws, size_t ws_size,
                              hipStream_t stream) {
  const float* x      = (const float*)d_in[0];
  const float* W_self = (const float*)d_in[1];
  const float* b_self = (const float*)d_in[2];
  const float* W_fwd  = (const float*)d_in[3];
  const float* b_fwd  = (const float*)d_in[4];
  const float* W_rev  = (const float*)d_in[5];
  const float* b_rev  = (const float*)d_in[6];
  const int*   dep    = (const int*)d_in[7];
  const int*   rel    = (const int*)d_in[8];
  const int*   gov    = (const int*)d_in[9];
  float* out = (float*)d_out;

  // workspace layout (xb/Wt/perm 256B-aligned; counter arrays only 64B-aligned)
  char* ws = (char*)d_ws;
  size_t o = 0;
  __bf16* xb = (__bf16*)(ws + o); o += (size_t)NN * DIM * 2;           // 15,360,000
  __bf16* Wt = (__bf16*)(ws + o); o += 21ull * DIM * DIM * 2;          //  2,752,512
  int* perm  = (int*)(ws + o);    o += (size_t)NE * 4;                 //    600,000
  int* cnt   = (int*)(ws + o);    o += 16 * 4;
  int* offs  = (int*)(ws + o);    o += 16 * 4;
  int* cur   = (int*)(ws + o);    o += 16 * 4;
  int* toff  = (int*)(ws + o);    o += 16 * 4;

  __bf16* Wt_self = Wt;
  __bf16* Wt_fwd  = Wt + 1ull  * DIM * DIM;
  __bf16* Wt_rev  = Wt + 11ull * DIM * DIM;

  // 1) precision conversion (x, weights transposed) into L2-resident bf16 copies
  cvt_x_bf16<<<(NN * DIM + 255) / 256, 256, 0, stream>>>(x, xb, NN * DIM);
  cvt_w_transpose<<<(1  * DIM * DIM + 255) / 256, 256, 0, stream>>>(W_self, Wt_self, 1);
  cvt_w_transpose<<<(NR * DIM * DIM + 255) / 256, 256, 0, stream>>>(W_fwd,  Wt_fwd,  NR);
  cvt_w_transpose<<<(NR * DIM * DIM + 255) / 256, 256, 0, stream>>>(W_rev,  Wt_rev,  NR);

  // 2) bucket edges by relation
  hipMemsetAsync(cnt, 0, 64 * 4, stream);   // zeroes cnt/offs/cur/toff
  rel_hist<<<(NE + 255) / 256, 256, 0, stream>>>(rel, cnt, NE);
  rel_scan<<<1, 1, 0, stream>>>(cnt, offs, toff);
  rel_fill<<<(NE + 255) / 256, 256, 0, stream>>>(rel, offs, cur, perm, NE);

  // 3) self transform writes d_out
  gemm_self<<<NN / 16, 256, 0, stream>>>(xb, Wt_self, b_self, out);

  // 4) per-relation gather-GEMM + atomic scatter-add (fwd then rev)
  const int max_tiles = NE / 16 + NR;
  gemm_edge<<<max_tiles, 256, 0, stream>>>(xb, Wt_fwd, b_fwd, perm, gov, dep, offs, toff, out);
  gemm_edge<<<max_tiles, 256, 0, stream>>>(xb, Wt_rev, b_rev, perm, dep, gov, offs, toff, out);
}
